// QuantizedLinear_52793738002579
// MI455X (gfx1250) — compile-verified, matrix operand order checked by reading, not run on hardware
//
#include <hip/hip_runtime.h>

typedef __attribute__((ext_vector_type(16))) _Float16 v16h;
typedef __attribute__((ext_vector_type(8)))  float    v8f;
typedef __attribute__((ext_vector_type(2)))  _Float16 half2v;

// Problem dims (fixed by the reference)
constexpr int TOKENS = 8192;
constexpr int D_IN   = 4096;
constexpr int D_OUT  = 4096;

// Tiling: block tile 128(M) x 256(N), K-step 32.
// 8 waves arranged 2(M) x 4(N); each wave owns a 64x64 C region
// = 4x4 grid of 16x16 WMMA tiles -> 16 v8f accumulators.
constexpr int BM = 128;
constexpr int BN = 256;
constexpr int BK = 32;
constexpr int NK = D_IN / BK;     // 128 K-steps

// LDS row stride: 32 f16 payload + 8 f16 pad = 40 halves = 80 bytes (16B aligned)
constexpr int LDS_STRIDE_H   = BK + 8;
constexpr int LDS_STRIDE_U32 = LDS_STRIDE_H / 2;   // 20 dwords

union Frag { uint4 q[2]; v16h v; };
union H2U  { half2v h; unsigned int u; };

__global__ __launch_bounds__(256)
void QuantizedLinear_52793738002579_kernel(const float* __restrict__ X,
                                           const float* __restrict__ W,
                                           const float* __restrict__ Bias,
                                           const float* __restrict__ Scale,
                                           float* __restrict__ Y) {
    __shared__ unsigned int xs[2][BM * LDS_STRIDE_U32];   // 20 KB
    __shared__ unsigned int ws[2][BN * LDS_STRIDE_U32];   // 40 KB

    const int tid      = threadIdx.x;
    const int lane     = tid & 31;
    const int wave     = tid >> 5;     // 0..7
    const int wm       = wave >> 2;    // 0..1  (M direction)
    const int wn       = wave & 3;     // 0..3  (N direction)
    const int lane16   = lane & 15;
    const int half_sel = lane >> 4;    // 0: K 0-7 / 16-23, 1: K 8-15 / 24-31

    const int block_m = blockIdx.y * BM;
    const int block_n = blockIdx.x * BN;

    v8f acc[4][4];
#pragma unroll
    for (int mi = 0; mi < 4; ++mi)
#pragma unroll
        for (int ni = 0; ni < 4; ++ni)
            acc[mi][ni] = (v8f)0.0f;

    const float scale = Scale[0];

    // Stage one K-step: 128x32 fp32 x-tile and 256x32 fp32 W-tile into LDS
    // buffer `buf`, converting to f16 in-flight.
    //   x: 1024 float4 / 256 thr = 4 each; W: 2048 float4 / 256 thr = 8 each.
    auto stage = [&](int kt, int buf) {
        const int k0 = kt * BK;
#pragma unroll
        for (int j = 0; j < 4; ++j) {
            const int idx = tid + j * 256;   // 0..1023
            const int row = idx >> 3;        // 0..127
            const int c4  = idx & 7;
            const float4 v = *(const float4*)(X + (size_t)(block_m + row) * D_IN + k0 + c4 * 4);
            H2U a, b;
            a.h.x = (_Float16)v.x;  a.h.y = (_Float16)v.y;
            b.h.x = (_Float16)v.z;  b.h.y = (_Float16)v.w;
            *(uint2*)&xs[buf][row * LDS_STRIDE_U32 + c4 * 2] = make_uint2(a.u, b.u);
        }
#pragma unroll
        for (int j = 0; j < 8; ++j) {
            const int idx = tid + j * 256;   // 0..2047
            const int row = idx >> 3;        // 0..255
            const int c4  = idx & 7;
            const float4 v = *(const float4*)(W + (size_t)(block_n + row) * D_IN + k0 + c4 * 4);
            H2U a, b;
            a.h.x = (_Float16)v.x;  a.h.y = (_Float16)v.y;
            b.h.x = (_Float16)v.z;  b.h.y = (_Float16)v.w;
            *(uint2*)&ws[buf][row * LDS_STRIDE_U32 + c4 * 2] = make_uint2(a.u, b.u);
        }
    };

    // One K=32 WMMA step for this wave's 64x64 C region from LDS buffer `buf`.
    auto compute = [&](int buf) {
        Frag afrag[4], bfrag[4];
#pragma unroll
        for (int mi = 0; mi < 4; ++mi) {
            const int row = wm * 64 + mi * 16 + lane16;
            const unsigned int* base = &xs[buf][row * LDS_STRIDE_U32 + half_sel * 4];
            afrag[mi].q[0] = *(const uint4*)(base);      // K 0-7   / 8-15
            afrag[mi].q[1] = *(const uint4*)(base + 8);  // K 16-23 / 24-31
        }
#pragma unroll
        for (int ni = 0; ni < 4; ++ni) {
            const int row = wn * 64 + ni * 16 + lane16;  // output column = W row
            const unsigned int* base = &ws[buf][row * LDS_STRIDE_U32 + half_sel * 4];
            bfrag[ni].q[0] = *(const uint4*)(base);
            bfrag[ni].q[1] = *(const uint4*)(base + 8);
        }
#pragma unroll
        for (int mi = 0; mi < 4; ++mi)
#pragma unroll
            for (int ni = 0; ni < 4; ++ni)
                acc[mi][ni] = __builtin_amdgcn_wmma_f32_16x16x32_f16(
                    /*neg_a=*/false, afrag[mi].v,
                    /*neg_b=*/false, bfrag[ni].v,
                    /*c_mod=*/(short)0, acc[mi][ni],
                    /*reuse_a=*/false, /*reuse_b=*/false);
    };

    stage(0, 0);
    __syncthreads();

    for (int kt = 0; kt < NK; ++kt) {
        const int cur = kt & 1;
        if (kt + 1 < NK) stage(kt + 1, cur ^ 1);
        compute(cur);
        __syncthreads();
    }

    // Epilogue: y = acc * scale + bias, streamed with non-temporal stores so
    // the 128 MB output does not evict the L2-resident x/W operands.
    // C/D layout: VGPR r holds rows (tile_m + r) for lanes 0-15 and
    // (tile_m + 8 + r) for lanes 16-31; N = lane16 within the 16-wide tile.
#pragma unroll
    for (int ni = 0; ni < 4; ++ni) {
        const int col = block_n + wn * 64 + ni * 16 + lane16;
        const float bias_v = Bias[col];
#pragma unroll
        for (int mi = 0; mi < 4; ++mi) {
            const int row0 = block_m + wm * 64 + mi * 16 + half_sel * 8;
#pragma unroll
            for (int r = 0; r < 8; ++r) {
                __builtin_nontemporal_store(acc[mi][ni][r] * scale + bias_v,
                                            Y + (size_t)(row0 + r) * D_OUT + col);
            }
        }
    }
}

extern "C" void kernel_launch(void* const* d_in, const int* in_sizes, int n_in,
                              void* d_out, int out_size, void* d_ws, size_t ws_size,
                              hipStream_t stream) {
    (void)in_sizes; (void)n_in; (void)d_ws; (void)ws_size; (void)out_size;
    const float* x     = (const float*)d_in[0];
    const float* w     = (const float*)d_in[1];
    const float* bias  = (const float*)d_in[2];
    const float* scale = (const float*)d_in[3];
    float* y           = (float*)d_out;

    dim3 grid(D_OUT / BN, TOKENS / BM);   // (16, 64)
    dim3 block(256);
    QuantizedLinear_52793738002579_kernel<<<grid, block, 0, stream>>>(x, w, bias, scale, y);
}
